// Block_1511828488339
// MI455X (gfx1250) — matrix operand order repaired and checked by reading, not compiled, not run on hardware
//
#include <hip/hip_runtime.h>
#include <math.h>

// ---------------------------------------------------------------------------
// Transformer block with linear attention (elu+1 feature map) for MI455X.
// Memory-bound problem (~1 GB traffic vs 0.41 TFLOP) -> bf16 activations +
// v_wmma_f32_16x16x32_bf16 everywhere, f32 accumulation.
// GEMM A-tiles are moved by the Tensor Data Mover (tensor_load_to_lds) with
// hardware LDS padding; B-tiles are transposed into LDS by the waves.
// ---------------------------------------------------------------------------

typedef __bf16 bf16_t;
typedef __attribute__((ext_vector_type(8)))  bf16_t v8bf;
typedef __attribute__((ext_vector_type(16))) bf16_t v16bf;
typedef __attribute__((ext_vector_type(8)))  float  v8f;
typedef __attribute__((ext_vector_type(4)))  unsigned int u32x4;
typedef __attribute__((ext_vector_type(8)))  int i32x8;
typedef __attribute__((ext_vector_type(4)))  int i32x4;

#define BATCH 4
#define SEQ   4096
#define HID   1024
#define NHEAD 16
#define HDIM  64
#define MLP   4096
#define ROWS  (BATCH * SEQ)   // 16384

#if __has_builtin(__builtin_amdgcn_tensor_load_to_lds) && __has_builtin(__builtin_amdgcn_s_wait_tensorcnt)
#define USE_TDM 1
#else
#define USE_TDM 0
#endif

union Frag16 { v16bf v; v8bf h[2]; };

// A-fragment (16x32 bf16, M x K): lane half selects K-offset 0/8; element
// pairs map VGPR0..3 -> K 0..7(+half*8), VGPR4..7 -> K 16..23(+half*8).
// p points at LDS row m, element (kbase + half*8); pitch keeps p 16B aligned.
__device__ __forceinline__ v16bf load_a_frag(const bf16_t* p) {
    Frag16 f;
    f.h[0] = *(const v8bf*)(p);        // K = k..k+7
    f.h[1] = *(const v8bf*)(p + 16);   // K = k+16..k+23
    return f.v;
}

// B-fragment (32x16 bf16, K x N): lane holds column n = lane&15; same K order
// as A: elements 0..7 -> K half*8+0..7, elements 8..15 -> K 16+half*8+0..7.
// B is staged K-contiguous per column (transposed in LDS).
__device__ __forceinline__ v16bf load_b_frag(const bf16_t* p) {
    Frag16 f;
    f.h[0] = *(const v8bf*)(p);        // K = k..k+7
    f.h[1] = *(const v8bf*)(p + 16);   // K = k+16..k+23
    return f.v;
}

__device__ __forceinline__ float gelu_tanh(float x) {
    float x3 = x * x * x;
    return 0.5f * x * (1.0f + tanhf(0.79788456080286535588f * (x + 0.044715f * x3)));
}
__device__ __forceinline__ float elu_plus1(float x) {
    return x > 0.0f ? x + 1.0f : expf(x);   // elu(x)+1
}

#if USE_TDM
// Tensor Data Mover: DMA a 128x64-bf16 row-major tile (rows stride K elems)
// into LDS at lds_off, padding each 32-DWORD row (64 bf16) with 4 DWORDs
// (8 bf16) so the landed pitch is 72 elements = 144B (matches load_a_frag).
// D# layout per CDNA5 ISA ch.8 (group0: count/lds/global addr/type=2;
// group1: data_size=2B, pad_interval code 4 (32 DW), pad_amount code 3 (4 DW),
// tensor_dim0=K, tensor_dim1=M, tile_dim0=64, tile_dim1=128, dim0_stride=K).
__device__ __forceinline__ void tdm_load_tile_a(const bf16_t* gptr,
                                                unsigned lds_off, int K, int M) {
    unsigned long long ga = (unsigned long long)(uintptr_t)gptr;
    u32x4 g0;
    g0.x = 1u;                                   // count=1, user descriptor
    g0.y = lds_off;                              // LDS byte address
    g0.z = (unsigned)ga;                         // global_addr[31:0]
    g0.w = (unsigned)((ga >> 32) & 0x01FFFFFFu)  // global_addr[56:32]
         | (2u << 30);                           // type = 2 ("image")
    i32x8 g1;
    g1[0] = (int)((1u << 16)     // data_size = 1 -> 2 bytes
                | (1u << 20)     // pad_enable
                | (4u << 22)     // pad_interval: 32 DWORDs
                | (3u << 25));   // pad_amount: 4 DWORDs
    g1[1] = (int)(((unsigned)K & 0xFFFFu) << 16);                    // tensor_dim0 lo
    g1[2] = (int)(((unsigned)K >> 16) | (((unsigned)M & 0xFFFFu) << 16)); // dim0 hi | dim1 lo
    g1[3] = (int)(((unsigned)M >> 16) | (64u << 16));                // dim1 hi | tile_dim0=64
    g1[4] = 128;                                                     // tile_dim1=128, tile_dim2=0
    g1[5] = K;                                                       // tensor_dim0_stride lo
    g1[6] = 0;                                                       // stride hi | dim1_stride lo
    g1[7] = 0;
    i32x4 z4 = {0, 0, 0, 0};
#if __clang_major__ >= 23
    i32x8 z8 = {0, 0, 0, 0, 0, 0, 0, 0};
    __builtin_amdgcn_tensor_load_to_lds(g0, g1, z4, z4, z8, 0);
#else
    __builtin_amdgcn_tensor_load_to_lds(g0, g1, z4, z4, 0);
#endif
}
#endif  // USE_TDM

// ---------------------------------------------------------------------------
// f32 -> bf16 cast (weights)
// ---------------------------------------------------------------------------
__global__ void cast_f32_bf16_kernel(const float* __restrict__ src,
                                     bf16_t* __restrict__ dst, int n) {
    int i = blockIdx.x * 256 + threadIdx.x;
    if (i < n) dst[i] = (bf16_t)src[i];
}

// ---------------------------------------------------------------------------
// LayerNorm over H=1024, output bf16. One 256-thread block per row.
// ---------------------------------------------------------------------------
__global__ __launch_bounds__(256) void layernorm_bf16_kernel(
    const float* __restrict__ x, const float* __restrict__ w,
    const float* __restrict__ b, bf16_t* __restrict__ out) {
    __shared__ float s1[256];
    __shared__ float s2[256];
    const int tid = threadIdx.x;
    const int row = blockIdx.x;
    const float* xr = x + (size_t)row * HID;

    float v[4], s = 0.0f, sq = 0.0f;
#pragma unroll
    for (int j = 0; j < 4; ++j) {
        float t = xr[tid + 256 * j];
        v[j] = t; s += t; sq += t * t;
    }
    s1[tid] = s; s2[tid] = sq;
    __syncthreads();
    for (int st = 128; st > 0; st >>= 1) {
        if (tid < st) { s1[tid] += s1[tid + st]; s2[tid] += s2[tid + st]; }
        __syncthreads();
    }
    const float mu  = s1[0] * (1.0f / HID);
    const float var = s2[0] * (1.0f / HID) - mu * mu;
    const float rs  = rsqrtf(var + 1e-5f);
#pragma unroll
    for (int j = 0; j < 4; ++j) {
        int i = tid + 256 * j;
        out[(size_t)row * HID + i] = (bf16_t)((v[j] - mu) * rs * w[i] + b[i]);
    }
}

// ---------------------------------------------------------------------------
// Tiled bf16 WMMA GEMM: C[M,N] = A[M,K] * W[K,N] + bias, epilogue EPI.
//   EPI 1: elu+1, store bf16 into head layout [B,NH,S,HD]   (Q, K proj)
//   EPI 2: plain, store bf16 into head layout               (V proj)
//   EPI 3: gelu,  store bf16 row-major                      (FC)
//   EPI 4: out_f32 = resid + acc + bias, row-major          (O-proj, Proj)
// Block tile 128x128, K-stage 64, 8 waves (4x2), wave tile 32x64 (2x4 WMMA).
// A tile arrives via TDM (hardware-padded pitch 72); B tile is transposed
// into LDS by the waves while the DMA is in flight.
// ---------------------------------------------------------------------------
template <int EPI>
__global__ __launch_bounds__(256) void gemm_bf16_kernel(
    const bf16_t* __restrict__ A, const bf16_t* __restrict__ W,
    const float* __restrict__ bias, const float* __restrict__ resid,
    bf16_t* __restrict__ out_bf, float* __restrict__ out_f,
    int M, int N, int K) {
    __shared__ bf16_t Alds[128 * 72];   // [m][k], pitch 72 -> 144B rows (16B aligned)
    __shared__ bf16_t Btr [128 * 80];   // [n][k], pitch 80 -> 160B rows (16B aligned)

    const int tid  = threadIdx.x;
    const int lane = tid & 31, wid = tid >> 5;
    const int half = lane >> 4, ln15 = lane & 15;
    const int waveM = wid & 3, waveN = wid >> 2;
    const int m0 = blockIdx.x * 128, n0 = blockIdx.y * 128;

    const v8f vzero = {0.f, 0.f, 0.f, 0.f, 0.f, 0.f, 0.f, 0.f};
    v8f acc[2][4];
#pragma unroll
    for (int i = 0; i < 2; ++i)
#pragma unroll
        for (int j = 0; j < 4; ++j) acc[i][j] = vzero;

#if USE_TDM
    const unsigned ldsA = (unsigned)(uintptr_t)(void*)Alds;  // LDS byte address
#endif

    for (int k0 = 0; k0 < K; k0 += 64) {
#if USE_TDM
        // One wave kicks off the A-tile DMA; it overlaps the B staging below.
        if (wid == 0)
            tdm_load_tile_a(A + (size_t)(m0)*K + k0, ldsA, K, M);
#else
        // stage A: 128x64 bf16, 8-element chunks, contiguous b128 LDS stores
#pragma unroll
        for (int i = 0; i < 4; ++i) {
            int c   = tid + 256 * i;
            int row = c >> 3, kc = (c & 7) * 8;
            v8bf av = *(const v8bf*)(A + (size_t)(m0 + row) * K + k0 + kc);
            *(v8bf*)(&Alds[row * 72 + kc]) = av;
        }
#endif
        // stage B transposed: 64x128 bf16 -> Btr[n][k]
#pragma unroll
        for (int i = 0; i < 4; ++i) {
            int c  = tid + 256 * i;
            int kr = c >> 4, nc = (c & 15) * 8;
            v8bf wv = *(const v8bf*)(W + (size_t)(k0 + kr) * N + n0 + nc);
#pragma unroll
            for (int j = 0; j < 8; ++j) Btr[(nc + j) * 80 + kr] = wv[j];
        }
        if (k0 + 64 < K)   // global_prefetch_b8 for next K-stage of W
            __builtin_prefetch(W + (size_t)(k0 + 64 + (tid >> 4)) * N + n0, 0, 0);
#if USE_TDM
        if (wid == 0) __builtin_amdgcn_s_wait_tensorcnt(0);
#endif
        __syncthreads();

#pragma unroll
        for (int kk = 0; kk < 64; kk += 32) {
            v16bf af[2], bf[4];
#pragma unroll
            for (int im = 0; im < 2; ++im)
                af[im] = load_a_frag(&Alds[(waveM * 32 + im * 16 + ln15) * 72 + kk + half * 8]);
#pragma unroll
            for (int in = 0; in < 4; ++in)
                bf[in] = load_b_frag(&Btr[(waveN * 64 + in * 16 + ln15) * 80 + kk + half * 8]);
#pragma unroll
            for (int im = 0; im < 2; ++im)
#pragma unroll
                for (int in = 0; in < 4; ++in)
                    acc[im][in] = __builtin_amdgcn_wmma_f32_16x16x32_bf16(
                        false, af[im], false, bf[in], (short)0, acc[im][in], false, false);
        }
        __syncthreads();
    }

    // Epilogue. C/D layout: VGPR i -> row i (lanes 0-15) / row 8+i (lanes 16-31),
    // col = lane&15 within the 16x16 tile.
#pragma unroll
    for (int im = 0; im < 2; ++im) {
#pragma unroll
        for (int in = 0; in < 4; ++in) {
#pragma unroll
            for (int i = 0; i < 8; ++i) {
                int m = m0 + waveM * 32 + im * 16 + half * 8 + i;
                int n = n0 + waveN * 64 + in * 16 + ln15;
                float val = acc[im][in][i] + bias[n];
                if constexpr (EPI == 1 || EPI == 2) {
                    int b = m >> 12, t = m & (SEQ - 1);
                    int h = n >> 6,  e = n & (HDIM - 1);
                    size_t idx = (((size_t)(b * NHEAD + h)) * SEQ + t) * HDIM + e;
                    out_bf[idx] = (bf16_t)(EPI == 1 ? elu_plus1(val) : val);
                } else if constexpr (EPI == 3) {
                    out_bf[(size_t)m * N + n] = (bf16_t)gelu_tanh(val);
                } else {  // EPI == 4
                    size_t idx = (size_t)m * N + n;
                    out_f[idx] = resid[idx] + val;
                }
            }
        }
    }
}

// ---------------------------------------------------------------------------
// Linear-attention state: kv[d][e] = sum_t kf[t,d]*v[t,e]; ksum[d] = sum_t kf.
// One block per (b,h). WMMA M=N=64, K=SEQ; both operands staged K-contiguous.
// ---------------------------------------------------------------------------
__global__ __launch_bounds__(256) void attn_kv_kernel(
    const bf16_t* __restrict__ kf, const bf16_t* __restrict__ vh,
    float* __restrict__ kv, float* __restrict__ ksum) {
    __shared__ bf16_t Ktr[64 * 72];   // [d][t]
    __shared__ bf16_t Vtr[64 * 72];   // [e][t]
    const int tid  = threadIdx.x;
    const int lane = tid & 31, wid = tid >> 5;
    const int half = lane >> 4, ln15 = lane & 15;
    const int waveM = wid & 3;    // d tile (4 x 16)
    const int waveN = wid >> 2;   // e tile pair (2 x 32)
    const int bh = blockIdx.x;
    const bf16_t* kbase = kf + (size_t)bh * SEQ * HDIM;
    const bf16_t* vbase = vh + (size_t)bh * SEQ * HDIM;

    const v8f vzero = {0.f, 0.f, 0.f, 0.f, 0.f, 0.f, 0.f, 0.f};
    v8f acc[2] = {vzero, vzero};
    float ks = 0.0f;

    for (int t0 = 0; t0 < SEQ; t0 += 64) {
        __syncthreads();
#pragma unroll
        for (int i = 0; i < 2; ++i) {
            int c = tid + 256 * i;
            int t = c >> 3, dc = (c & 7) * 8;
            v8bf kvv = *(const v8bf*)(kbase + (size_t)(t0 + t) * HDIM + dc);
            v8bf vv  = *(const v8bf*)(vbase + (size_t)(t0 + t) * HDIM + dc);
#pragma unroll
            for (int j = 0; j < 8; ++j) {
                Ktr[(dc + j) * 72 + t] = kvv[j];
                Vtr[(dc + j) * 72 + t] = vv[j];
            }
        }
        __syncthreads();

        if (tid < 64) {   // ksum partial over this t-tile (d = tid)
#pragma unroll
            for (int t = 0; t < 64; ++t) ks += (float)Ktr[tid * 72 + t];
        }
#pragma unroll
        for (int kk = 0; kk < 64; kk += 32) {
            v16bf af = load_a_frag(&Ktr[(waveM * 16 + ln15) * 72 + kk + half * 8]);
#pragma unroll
            for (int j = 0; j < 2; ++j) {
                v16bf bfj = load_b_frag(&Vtr[((waveN * 2 + j) * 16 + ln15) * 72 + kk + half * 8]);
                acc[j] = __builtin_amdgcn_wmma_f32_16x16x32_bf16(
                    false, af, false, bfj, (short)0, acc[j], false, false);
            }
        }
    }

#pragma unroll
    for (int j = 0; j < 2; ++j)
#pragma unroll
        for (int i = 0; i < 8; ++i) {
            int d = waveM * 16 + half * 8 + i;
            int e = (waveN * 2 + j) * 16 + ln15;
            kv[(size_t)bh * (HDIM * HDIM) + d * HDIM + e] = acc[j][i];
        }
    if (tid < 64) ksum[bh * HDIM + tid] = ks;
}

// ---------------------------------------------------------------------------
// norm[bh,t] = 1 / (qf[bh,t,:] . ksum[bh,:] + 1e-6)
// ---------------------------------------------------------------------------
__global__ void attn_norm_kernel(const bf16_t* __restrict__ qf,
                                 const float* __restrict__ ksum,
                                 float* __restrict__ nrm) {
    int gid = blockIdx.x * 256 + threadIdx.x;   // < B*NH*S
    int bh  = gid >> 12;
    const bf16_t* qr = qf + (size_t)gid * HDIM;
    const float*  ks = ksum + bh * HDIM;
    float s = 0.0f;
#pragma unroll
    for (int i = 0; i < HDIM; ++i) s += (float)qr[i] * ks[i];
    nrm[gid] = 1.0f / (s + 1e-6f);
}

// ---------------------------------------------------------------------------
// out[t,e] = (qf[t,:] . kv[:,e]) * norm[t], written to [B,S,H] bf16.
// Grid (SEQ/128, B*NH); wave w owns one 16-row t-tile x all 4 e-tiles.
// ---------------------------------------------------------------------------
__global__ __launch_bounds__(256) void attn_out_kernel(
    const bf16_t* __restrict__ qf, const float* __restrict__ kv,
    const float* __restrict__ nrm, bf16_t* __restrict__ attn) {
    __shared__ bf16_t Qlds[128 * 72];   // [t][d]
    __shared__ bf16_t KVtr[64 * 72];    // [e][d]
    const int tid  = threadIdx.x;
    const int lane = tid & 31, wid = tid >> 5;
    const int half = lane >> 4, ln15 = lane & 15;
    const int bh = blockIdx.y, b = bh >> 4, h = bh & (NHEAD - 1);
    const int t0 = blockIdx.x * 128;
    const bf16_t* qbase = qf + ((size_t)bh * SEQ + t0) * HDIM;
    const float*  kvb   = kv + (size_t)bh * (HDIM * HDIM);

#pragma unroll
    for (int i = 0; i < 4; ++i) {
        int c = tid + 256 * i;
        int t = c >> 3, dc = (c & 7) * 8;
        *(v8bf*)(&Qlds[t * 72 + dc]) = *(const v8bf*)(qbase + (size_t)t * HDIM + dc);
    }
#pragma unroll
    for (int i = 0; i < 16; ++i) {
        int idx = tid + 256 * i;
        int d = idx >> 6, e = idx & 63;
        KVtr[e * 72 + d] = (bf16_t)kvb[idx];
    }
    __syncthreads();

    const v8f vzero = {0.f, 0.f, 0.f, 0.f, 0.f, 0.f, 0.f, 0.f};
    v8f acc[4] = {vzero, vzero, vzero, vzero};
#pragma unroll
    for (int kk = 0; kk < 64; kk += 32) {
        v16bf af = load_a_frag(&Qlds[(wid * 16 + ln15) * 72 + kk + half * 8]);
#pragma unroll
        for (int j = 0; j < 4; ++j) {
            v16bf bfj = load_b_frag(&KVtr[(j * 16 + ln15) * 72 + kk + half * 8]);
            acc[j] = __builtin_amdgcn_wmma_f32_16x16x32_bf16(
                false, af, false, bfj, (short)0, acc[j], false, false);
        }
    }
#pragma unroll
    for (int j = 0; j < 4; ++j)
#pragma unroll
        for (int i = 0; i < 8; ++i) {
            int t = t0 + wid * 16 + half * 8 + i;
            float w = nrm[bh * SEQ + t];
            int e = j * 16 + ln15;
            attn[((size_t)b * SEQ + t) * HID + h * HDIM + e] = (bf16_t)(acc[j][i] * w);
        }
}

// ---------------------------------------------------------------------------
// Host launch
// ---------------------------------------------------------------------------
extern "C" void kernel_launch(void* const* d_in, const int* in_sizes, int n_in,
                              void* d_out, int out_size, void* d_ws, size_t ws_size,
                              hipStream_t stream) {
    (void)in_sizes; (void)n_in; (void)out_size; (void)ws_size;

    const float* x     = (const float*)d_in[0];
    const float* ln1_w = (const float*)d_in[1];
    const float* ln1_b = (const float*)d_in[2];
    const float* q_w   = (const float*)d_in[3];
    const float* q_b   = (const float*)d_in[4];
    const float* k_w   = (const float*)d_in[5];
    const float* k_b   = (const float*)d_in[6];
    const float* v_w   = (const float*)d_in[7];
    const float* v_b   = (const float*)d_in[8];
    const float* o_w   = (const float*)d_in[9];
    const float* o_b   = (const float*)d_in[10];
    const float* ln2_w = (const float*)d_in[11];
    const float* ln2_b = (const float*)d_in[12];
    const float* fc_w  = (const float*)d_in[13];
    const float* fc_b  = (const float*)d_in[14];
    const float* pj_w  = (const float*)d_in[15];
    const float* pj_b  = (const float*)d_in[16];
    float* out = (float*)d_out;

    char* p = (char*)d_ws;
    auto alloc = [&](size_t bytes) -> char* {
        char* r = p;
        p += (bytes + 255) & ~(size_t)255;
        return r;
    };

    const size_t HH   = (size_t)HID * HID;           // 1M
    const size_t HM   = (size_t)HID * MLP;           // 4M
    const size_t ACT  = (size_t)ROWS * HID;          // 16M
    const size_t ACTM = (size_t)ROWS * MLP;          // 64M

    bf16_t* Wq   = (bf16_t*)alloc(HH * 2);
    bf16_t* Wk   = (bf16_t*)alloc(HH * 2);
    bf16_t* Wv   = (bf16_t*)alloc(HH * 2);
    bf16_t* Wo   = (bf16_t*)alloc(HH * 2);
    bf16_t* Wfc  = (bf16_t*)alloc(HM * 2);
    bf16_t* Wpj  = (bf16_t*)alloc(HM * 2);
    bf16_t* ln1o = (bf16_t*)alloc(ACT * 2);
    bf16_t* qfb  = (bf16_t*)alloc(ACT * 2);          // [B,NH,S,HD]
    bf16_t* kfb  = (bf16_t*)alloc(ACT * 2);
    bf16_t* vhb  = (bf16_t*)alloc(ACT * 2);
    float*  kvb  = (float*) alloc((size_t)BATCH * NHEAD * HDIM * HDIM * 4);
    float*  ksum = (float*) alloc((size_t)BATCH * NHEAD * HDIM * 4);
    float*  nrm  = (float*) alloc((size_t)BATCH * NHEAD * SEQ * 4);
    bf16_t* attn = (bf16_t*)alloc(ACT * 2);          // [B,S,H]
    float*  x1   = (float*) alloc(ACT * 4);
    bf16_t* ln2o = (bf16_t*)alloc(ACT * 2);
    bf16_t* hmlp = (bf16_t*)alloc(ACTM * 2);

    auto cast = [&](const float* s, bf16_t* d, size_t n) {
        cast_f32_bf16_kernel<<<(int)((n + 255) / 256), 256, 0, stream>>>(s, d, (int)n);
    };
    cast(q_w, Wq, HH);  cast(k_w, Wk, HH);  cast(v_w, Wv, HH);
    cast(o_w, Wo, HH);  cast(fc_w, Wfc, HM); cast(pj_w, Wpj, HM);

    // LN1 -> bf16
    layernorm_bf16_kernel<<<ROWS, 256, 0, stream>>>(x, ln1_w, ln1_b, ln1o);

    // Q/K/V projections: elu+1 on Q,K; head layout output
    dim3 gHH(ROWS / 128, HID / 128);
    gemm_bf16_kernel<1><<<gHH, 256, 0, stream>>>(ln1o, Wq, q_b, nullptr, qfb, nullptr, ROWS, HID, HID);
    gemm_bf16_kernel<1><<<gHH, 256, 0, stream>>>(ln1o, Wk, k_b, nullptr, kfb, nullptr, ROWS, HID, HID);
    gemm_bf16_kernel<2><<<gHH, 256, 0, stream>>>(ln1o, Wv, v_b, nullptr, vhb, nullptr, ROWS, HID, HID);

    // linear attention
    attn_kv_kernel<<<BATCH * NHEAD, 256, 0, stream>>>(kfb, vhb, kvb, ksum);
    attn_norm_kernel<<<(BATCH * NHEAD * SEQ) / 256, 256, 0, stream>>>(qfb, ksum, nrm);
    attn_out_kernel<<<dim3(SEQ / 128, BATCH * NHEAD), 256, 0, stream>>>(qfb, kvb, nrm, attn);

    // O-projection + residual -> x1 (f32)
    gemm_bf16_kernel<4><<<gHH, 256, 0, stream>>>(attn, Wo, o_b, x, nullptr, x1, ROWS, HID, HID);

    // LN2 -> bf16
    layernorm_bf16_kernel<<<ROWS, 256, 0, stream>>>(x1, ln2_w, ln2_b, ln2o);

    // MLP: fc + gelu, then proj + residual -> out (f32)
    dim3 gFC(ROWS / 128, MLP / 128);
    gemm_bf16_kernel<3><<<gFC, 256, 0, stream>>>(ln2o, Wfc, fc_b, nullptr, hmlp, nullptr, ROWS, MLP, HID);
    gemm_bf16_kernel<4><<<gHH, 256, 0, stream>>>(hmlp, Wpj, pj_b, x1, nullptr, out, ROWS, HID, MLP);
}
